// WipModule_21990232556153
// MI455X (gfx1250) — matrix operand order repaired and checked by reading, not compiled, not run on hardware
//
#include <hip/hip_runtime.h>
#include <hip/hip_bf16.h>
#include <stdint.h>

typedef __attribute__((ext_vector_type(16))) __bf16   bf16x16;
typedef __attribute__((ext_vector_type(8)))  float    f32x8;
typedef __attribute__((ext_vector_type(4)))  int      i32x4;
typedef __attribute__((ext_vector_type(4)))  unsigned u32x4;
typedef __attribute__((ext_vector_type(8)))  unsigned u32x8;

// ---------------- geometry ----------------
constexpr int H = 64, W = 128, NPIX = H * W;       // full res
constexpr int H2 = 32, W2 = 64, NPIX2 = H2 * W2;   // half res
constexpr int ND = 169;                            // 13x13 displacements
constexpr int BBLK = 512;                          // packed-B block: 32 lanes x 16 bf16

// ---------------- workspace layout (bytes) ----------------
constexpr size_t OFF_FM1  = 0;
constexpr size_t SZ_FM    = (size_t)NPIX * 32 * 2;
constexpr size_t OFF_FM2  = OFF_FM1 + SZ_FM;
constexpr size_t OFF_SUM  = OFF_FM2 + SZ_FM;
constexpr size_t SZ_SUM   = (size_t)NPIX * 4;
constexpr size_t OFF_ZP   = OFF_SUM + SZ_SUM;          // 512B zero page
constexpr size_t SZ_ZP    = 512;
constexpr size_t OFF_W1   = OFF_ZP + SZ_ZP;            // 6 ntiles x 18 kb
constexpr size_t SZ_W1    = (size_t)6 * 18 * BBLK * 2;
constexpr size_t OFF_W2   = OFF_W1 + SZ_W1;            // 8 x 27
constexpr size_t SZ_W2    = (size_t)8 * 27 * BBLK * 2;
constexpr size_t OFF_W3   = OFF_W2 + SZ_W2;            // 8 x 36
constexpr size_t SZ_W3    = (size_t)8 * 36 * BBLK * 2;
constexpr size_t OFF_W4   = OFF_W3 + SZ_W3;            // 4 x 36
constexpr size_t SZ_W4    = (size_t)4 * 36 * BBLK * 2;
constexpr size_t OFF_WD   = OFF_W4 + SZ_W4;            // 4 parity x 2 x 8
constexpr size_t SZ_WD    = (size_t)4 * 2 * 8 * BBLK * 2;
constexpr size_t OFF_W5   = OFF_WD + SZ_WD;            // 1 x 9
constexpr size_t SZ_W5    = (size_t)1 * 9 * BBLK * 2;
constexpr size_t OFF_ACT1 = (OFF_W5 + SZ_W5 + 255) & ~(size_t)255;
constexpr size_t SZ_ACT1  = (size_t)ND * NPIX  *  96 * 2;
constexpr size_t OFF_ACT2 = OFF_ACT1 + SZ_ACT1;
constexpr size_t SZ_ACT2  = (size_t)ND * NPIX2 * 128 * 2;
constexpr size_t OFF_ACT3 = OFF_ACT2 + SZ_ACT2;
constexpr size_t OFF_ACT4 = OFF_ACT3 + SZ_ACT2;
constexpr size_t SZ_ACT4  = (size_t)ND * NPIX2 *  64 * 2;
constexpr size_t OFF_ACT5 = OFF_ACT4 + SZ_ACT4;

// ---------------- Tensor Data Mover: 2D tile (4B elems) global -> LDS ----------------
// rows x tile0_elems tile, global row stride = row_stride_elems, packed dense into LDS.
__device__ __forceinline__ void tdm_load_2d(unsigned lds_off, const void* gsrc,
                                            unsigned tile0_elems, unsigned rows,
                                            unsigned row_stride_elems) {
  unsigned long long ga = (unsigned long long)(uintptr_t)gsrc;
  u32x4 g0;
  g0[0] = 1u;                                   // count = 1 valid descriptor
  g0[1] = lds_off;                              // LDS byte address
  g0[2] = (unsigned)ga;                         // global_addr[31:0]
  g0[3] = (unsigned)(ga >> 32) | 0x80000000u;   // global_addr[56:32] | type=2<<30
  u32x8 g1;
  g1[0] = 2u << 16;                                         // data_size = 4B
  g1[1] = (tile0_elems & 0xffffu) << 16;                    // tensor_dim0[15:0]
  g1[2] = (tile0_elems >> 16) | ((rows & 0xffffu) << 16);   // td0 hi | tensor_dim1 lo
  g1[3] = (rows >> 16) | ((tile0_elems & 0xffffu) << 16);   // td1 hi | tile_dim0
  g1[4] = rows & 0xffffu;                                   // tile_dim1 (tile_dim2=0)
  g1[5] = row_stride_elems;                                 // tensor_dim0_stride[31:0]
  g1[6] = 0u;
  g1[7] = 0u;
  asm volatile("tensor_load_to_lds %0, %1" :: "s"(g0), "s"(g1) : "memory");
}

// ---------------- prep: channels-last bf16 maps + channel sums + zero page ----------------
__global__ void prep_maps(const float* __restrict__ f1, const float* __restrict__ f2,
                          __bf16* __restrict__ fm1, __bf16* __restrict__ fm2,
                          float* __restrict__ fsum, int* __restrict__ zp) {
  int p = blockIdx.x * blockDim.x + threadIdx.x;
  if (blockIdx.x == 0 && threadIdx.x < 128) zp[threadIdx.x] = 0;   // 512B zeros
  if (p >= NPIX) return;
  float s = 0.f;
#pragma unroll
  for (int c = 0; c < 32; ++c) {
    float a = f1[(size_t)c * NPIX + p];
    float b = f2[(size_t)c * NPIX + p];
    fm1[(size_t)p * 32 + c] = (__bf16)a;
    fm2[(size_t)p * 32 + c] = (__bf16)b;
    s += b;
  }
  fsum[p] = s;
}

// ---------------- pack 3x3 conv weights (OIHW f32) into WMMA-B blocks ----------------
// Block (nt*KB + kb): lane L holds N = nt*16+L%16, element e=2v+h holds K = kb*32+16*(e/8)+8*(L/16)+e%8.
__global__ void pack_conv_w(const float* __restrict__ w, __bf16* __restrict__ dst,
                            int Cout, int Cin, int KB, int ntiles) {
  int idx = blockIdx.x * blockDim.x + threadIdx.x;
  int total = ntiles * KB * 256;
  if (idx >= total) return;
  int v    = idx & 7;
  int lane = (idx >> 3) & 31;
  int blk  = idx >> 8;
  int kb   = blk % KB;
  int nt   = blk / KB;
  int n = nt * 16 + (lane & 15);
#pragma unroll
  for (int hh = 0; hh < 2; ++hh) {
    int e = 2 * v + hh;
    int k = kb * 32 + ((e >> 3) << 4) + ((lane >> 4) << 3) + (e & 7);
    int tap = k / Cin, c = k % Cin;
    int ky = tap / 3, kx = tap % 3;
    float val = (n < Cout) ? w[(((size_t)n * Cin + c) * 3 + ky) * 3 + kx] : 0.f;
    dst[(size_t)idx * 2 + hh] = (__bf16)val;
  }
}

// ---------------- pack 4x4 deconv weights into 4 parity variants ----------------
__global__ void pack_deconv_w(const float* __restrict__ wd, __bf16* __restrict__ dst) {
  int idx = blockIdx.x * blockDim.x + threadIdx.x;   // 4*2*8*256 = 16384
  if (idx >= 16384) return;
  int v    = idx & 7;
  int lane = (idx >> 3) & 31;
  int kb   = (idx >> 8) & 7;
  int nt   = (idx >> 11) & 1;
  int par  = idx >> 12;
  int py = par >> 1, px = par & 1;
  int o = nt * 16 + (lane & 15);
#pragma unroll
  for (int hh = 0; hh < 2; ++hh) {
    int e = 2 * v + hh;
    int k = kb * 32 + ((e >> 3) << 4) + ((lane >> 4) << 3) + (e & 7);
    int t = k >> 6, c = k & 63;
    int a = t >> 1, b = t & 1;
    int ky = 2 * a + py, kx = 2 * b + px;
    float val = wd[(((size_t)o * 64 + c) * 4 + ky) * 4 + kx];
    dst[(size_t)idx * 2 + hh] = (__bf16)val;
  }
}

// ---------------- conv1: mvol construction fused into implicit-GEMM WMMA ----------------
// Wave computes 16 pixels x 32 channels (2 N-tiles); K = 9 taps * 64 ch.
__global__ __launch_bounds__(256, 2) void conv1_mvol_wmma(
    const __bf16* __restrict__ fm1, const __bf16* __restrict__ fm2,
    const float* __restrict__ fsum, const __bf16* __restrict__ Bpack,
    __bf16* __restrict__ act1, const __bf16* __restrict__ zp) {
  constexpr int KB = 18, NT2 = 2;
  __shared__ alignas(32) int Bsh[NT2 * KB * 256];   // 36KB
  const int lane = threadIdx.x & 31;
  const int wv   = threadIdx.x >> 5;
  const int pair  = blockIdx.y;                     // of 3 (96 out ch)
  const int mtile = blockIdx.x * 8 + wv;
  const int d = blockIdx.z;
  const int sy = (d % 13) - 6;
  const int sx = (d / 13) - 6;

  if (wv == 0) {
    tdm_load_2d((unsigned)(uintptr_t)Bsh,
                Bpack + (size_t)pair * NT2 * KB * BBLK,
                KB * 256, NT2, KB * 256);
    __builtin_amdgcn_s_wait_tensorcnt(0);
  }
  __syncthreads();

  const int m   = lane & 15;
  const int klo = (lane >> 4) * 8;
  const int p = mtile * 16 + m;
  const int y = p >> 7, x = p & 127;

  // hoist per-tap validity + source pointers (zero page when invalid)
  const __bf16* pf[9][2];
#pragma unroll
  for (int t = 0; t < 9; ++t) {
    int qy = y + t / 3 - 1, qx = x + t % 3 - 1;
    int ty = qy + sy, tx = qx + sx;
    bool val = ((unsigned)qy < (unsigned)H) && ((unsigned)qx < (unsigned)W) &&
               ((unsigned)ty < (unsigned)H) && ((unsigned)tx < (unsigned)W);
    float sv = fsum[val ? ty * W + tx : 0];
    val = val && (sv != 0.f);
    pf[t][0] = val ? fm1 + (size_t)(qy * W + qx) * 32 : zp;
    pf[t][1] = val ? fm2 + (size_t)(ty * W + tx) * 32 : zp;
  }

  f32x8 acc[NT2] = {};
#pragma unroll
  for (int kb = 0; kb < KB; ++kb) {
    const int tap = kb >> 1;
    const __bf16* src = pf[tap][kb & 1] + klo;
    union { i32x4 q[2]; bf16x16 a; } au;
    au.q[0] = *(const i32x4*)(src);
    au.q[1] = *(const i32x4*)(src + 16);
#pragma unroll
    for (int r = 0; r < NT2; ++r) {
      bf16x16 bfrag = *(const bf16x16*)((const __bf16*)Bsh + (r * KB + kb) * BBLK + lane * 16);
      acc[r] = __builtin_amdgcn_wmma_f32_16x16x32_bf16(false, au.a, false, bfrag,
                                                       (short)0, acc[r], false, false);
    }
  }

  const int nn  = lane & 15;
  const int mhi = (lane >> 4) * 8;
  __bf16* out = act1 + (size_t)d * NPIX * 96;
#pragma unroll
  for (int r = 0; r < NT2; ++r)
#pragma unroll
    for (int v = 0; v < 8; ++v) {
      int pp = mtile * 16 + mhi + v;
      out[(size_t)pp * 96 + (pair * NT2 + r) * 16 + nn] = (__bf16)fmaxf(acc[r][v], 0.f);
    }
}

// ---------------- generic 3x3 conv, implicit GEMM on WMMA bf16 ----------------
template <int CIN, int COUT, int STRIDE, int HIN, int WIN, bool RELU, bool LAST, int NT2>
__global__ __launch_bounds__(256, 2) void conv3x3_wmma(
    const __bf16* __restrict__ actIn, const __bf16* __restrict__ Bpack,
    __bf16* __restrict__ actOut, float* __restrict__ outF32,
    const __bf16* __restrict__ zp) {
  constexpr int KB  = 9 * CIN / 32;
  constexpr int KBC = (KB % 18 == 0) ? 18 : (KB % 9 == 0) ? 9 : KB;
  constexpr int CH  = KB / KBC;
  constexpr int HOUT = HIN / STRIDE, WOUT = WIN / STRIDE;
  __shared__ alignas(32) int Bsh[NT2 * KBC * 256];

  const int lane = threadIdx.x & 31;
  const int wv   = threadIdx.x >> 5;
  const int pair  = blockIdx.y;
  const int mtile = blockIdx.x * 8 + wv;
  const int d = blockIdx.z;

  const int m   = lane & 15;
  const int klo = (lane >> 4) * 8;
  const int p = mtile * 16 + m;
  const int y = p / WOUT, x = p % WOUT;
  const __bf16* in = actIn + (size_t)d * HIN * WIN * CIN;

  const __bf16* aptr[9];
#pragma unroll
  for (int t = 0; t < 9; ++t) {
    int iy = y * STRIDE + t / 3 - 1, ix = x * STRIDE + t % 3 - 1;
    bool inb = ((unsigned)iy < (unsigned)HIN) && ((unsigned)ix < (unsigned)WIN);
    aptr[t] = inb ? (in + (size_t)(iy * WIN + ix) * CIN) : zp;
  }

  f32x8 acc[NT2] = {};
#pragma unroll
  for (int ch = 0; ch < CH; ++ch) {
    if (ch) __syncthreads();
    if (wv == 0) {
      tdm_load_2d((unsigned)(uintptr_t)Bsh,
                  Bpack + ((size_t)pair * NT2 * KB + ch * KBC) * BBLK,
                  KBC * 256, NT2, KB * 256);
      __builtin_amdgcn_s_wait_tensorcnt(0);
    }
    __syncthreads();
#pragma unroll
    for (int j = 0; j < KBC; ++j) {
      const int kb = ch * KBC + j;
      const int tap = (kb * 32) / CIN, cb = (kb * 32) % CIN;
      const __bf16* src = aptr[tap] + cb + klo;
      union { i32x4 q[2]; bf16x16 a; } au;
      au.q[0] = *(const i32x4*)(src);
      au.q[1] = *(const i32x4*)(src + 16);
#pragma unroll
      for (int r = 0; r < NT2; ++r) {
        bf16x16 bfrag = *(const bf16x16*)((const __bf16*)Bsh + (r * KBC + j) * BBLK + lane * 16);
        acc[r] = __builtin_amdgcn_wmma_f32_16x16x32_bf16(false, au.a, false, bfrag,
                                                         (short)0, acc[r], false, false);
      }
    }
  }

  const int nn  = lane & 15;
  const int mhi = (lane >> 4) * 8;
  if (LAST) {
    float* o = outF32 + (size_t)d * (HOUT * WOUT);
    if (nn == 0) {
#pragma unroll
      for (int v = 0; v < 8; ++v) o[mtile * 16 + mhi + v] = acc[0][v];
    }
  } else {
    __bf16* o = actOut + (size_t)d * (HOUT * WOUT) * COUT;
#pragma unroll
    for (int r = 0; r < NT2; ++r)
#pragma unroll
      for (int v = 0; v < 8; ++v) {
        int pp = mtile * 16 + mhi + v;
        float rr = RELU ? fmaxf(acc[r][v], 0.f) : acc[r][v];
        o[(size_t)pp * COUT + (pair * NT2 + r) * 16 + nn] = (__bf16)rr;
      }
  }
}

// ---------------- 4x4 stride-2 transpose conv as 4 parity-subgrid GEMMs ----------------
__global__ __launch_bounds__(256, 2) void deconv_wmma(
    const __bf16* __restrict__ act4, const __bf16* __restrict__ Bpack,
    __bf16* __restrict__ act5, const __bf16* __restrict__ zp) {
  constexpr int KB = 8, NT2 = 2;
  __shared__ alignas(32) int Bsh[NT2 * KB * 256];   // 16KB
  const int lane = threadIdx.x & 31;
  const int wv   = threadIdx.x >> 5;
  const int parity = blockIdx.y;                    // (py,px)
  const int py = parity >> 1, px = parity & 1;
  const int mtile = blockIdx.x * 8 + wv;
  const int d = blockIdx.z;

  if (wv == 0) {
    tdm_load_2d((unsigned)(uintptr_t)Bsh,
                Bpack + (size_t)parity * NT2 * KB * BBLK,
                KB * 256, NT2, KB * 256);
    __builtin_amdgcn_s_wait_tensorcnt(0);
  }
  __syncthreads();

  const int m   = lane & 15;
  const int klo = (lane >> 4) * 8;
  const int p = mtile * 16 + m;
  const int Y = p / W2, X = p % W2;                 // 32x64 parity subgrid
  const __bf16* in = act4 + (size_t)d * NPIX2 * 64;

  const __bf16* aptr[4];
#pragma unroll
  for (int t = 0; t < 4; ++t) {
    int iy = Y + py + (t >> 1) - 1, ix = X + px + (t & 1) - 1;
    bool inb = ((unsigned)iy < (unsigned)H2) && ((unsigned)ix < (unsigned)W2);
    aptr[t] = inb ? (in + (size_t)(iy * W2 + ix) * 64) : zp;
  }

  f32x8 acc[NT2] = {};
#pragma unroll
  for (int kb = 0; kb < KB; ++kb) {
    const int tap = kb >> 1;
    const int cb = (kb & 1) * 32;
    const __bf16* src = aptr[tap] + cb + klo;
    union { i32x4 q[2]; bf16x16 a; } au;
    au.q[0] = *(const i32x4*)(src);
    au.q[1] = *(const i32x4*)(src + 16);
#pragma unroll
    for (int r = 0; r < NT2; ++r) {
      bf16x16 bfrag = *(const bf16x16*)((const __bf16*)Bsh + (r * KB + kb) * BBLK + lane * 16);
      acc[r] = __builtin_amdgcn_wmma_f32_16x16x32_bf16(false, au.a, false, bfrag,
                                                       (short)0, acc[r], false, false);
    }
  }

  const int nn  = lane & 15;
  const int mhi = (lane >> 4) * 8;
  __bf16* o = act5 + (size_t)d * NPIX * 32;
#pragma unroll
  for (int r = 0; r < NT2; ++r)
#pragma unroll
    for (int v = 0; v < 8; ++v) {
      int pp = mtile * 16 + mhi + v;
      int oy = 2 * (pp / W2) + py, ox = 2 * (pp % W2) + px;
      o[(size_t)(oy * W + ox) * 32 + r * 16 + nn] = (__bf16)fmaxf(acc[r][v], 0.f);
    }
}

// ---------------- host-side launch ----------------
extern "C" void kernel_launch(void* const* d_in, const int* in_sizes, int n_in,
                              void* d_out, int out_size, void* d_ws, size_t ws_size,
                              hipStream_t stream) {
  (void)in_sizes; (void)n_in; (void)out_size; (void)ws_size;
  const float* fmap1 = (const float*)d_in[0];
  const float* fmap2 = (const float*)d_in[1];
  const float* w1 = (const float*)d_in[2];
  const float* w2 = (const float*)d_in[3];
  const float* w3 = (const float*)d_in[4];
  const float* w4 = (const float*)d_in[5];
  const float* wd = (const float*)d_in[6];
  const float* w5 = (const float*)d_in[7];

  char* ws = (char*)d_ws;
  __bf16* FM1  = (__bf16*)(ws + OFF_FM1);
  __bf16* FM2  = (__bf16*)(ws + OFF_FM2);
  float*  FSUM = (float*)(ws + OFF_SUM);
  __bf16* ZP   = (__bf16*)(ws + OFF_ZP);
  __bf16* W1   = (__bf16*)(ws + OFF_W1);
  __bf16* W2   = (__bf16*)(ws + OFF_W2);
  __bf16* W3   = (__bf16*)(ws + OFF_W3);
  __bf16* W4   = (__bf16*)(ws + OFF_W4);
  __bf16* WD   = (__bf16*)(ws + OFF_WD);
  __bf16* W5   = (__bf16*)(ws + OFF_W5);
  __bf16* ACT1 = (__bf16*)(ws + OFF_ACT1);
  __bf16* ACT2 = (__bf16*)(ws + OFF_ACT2);
  __bf16* ACT3 = (__bf16*)(ws + OFF_ACT3);
  __bf16* ACT4 = (__bf16*)(ws + OFF_ACT4);
  __bf16* ACT5 = (__bf16*)(ws + OFF_ACT5);

  prep_maps<<<(NPIX + 255) / 256, 256, 0, stream>>>(fmap1, fmap2, FM1, FM2, FSUM, (int*)ZP);
  pack_conv_w<<<6 * 18, 256, 0, stream>>>(w1, W1, 96, 64, 18, 6);
  pack_conv_w<<<8 * 27, 256, 0, stream>>>(w2, W2, 128, 96, 27, 8);
  pack_conv_w<<<8 * 36, 256, 0, stream>>>(w3, W3, 128, 128, 36, 8);
  pack_conv_w<<<4 * 36, 256, 0, stream>>>(w4, W4, 64, 128, 36, 4);
  pack_conv_w<<<1 * 9, 256, 0, stream>>>(w5, W5, 1, 32, 9, 1);
  pack_deconv_w<<<64, 256, 0, stream>>>(wd, WD);

  // conv1 (fused mvol): M=8192, N=96 -> 64 mgroups x 3 pairs x 169
  conv1_mvol_wmma<<<dim3(64, 3, ND), 256, 0, stream>>>(FM1, FM2, FSUM, W1, ACT1, ZP);
  // conv2: 96->128 stride 2
  conv3x3_wmma<96, 128, 2, 64, 128, true, false, 2>
      <<<dim3(16, 4, ND), 256, 0, stream>>>(ACT1, W2, ACT2, nullptr, ZP);
  // conv3: 128->128
  conv3x3_wmma<128, 128, 1, 32, 64, true, false, 2>
      <<<dim3(16, 4, ND), 256, 0, stream>>>(ACT2, W3, ACT3, nullptr, ZP);
  // conv4: 128->64
  conv3x3_wmma<128, 64, 1, 32, 64, true, false, 2>
      <<<dim3(16, 2, ND), 256, 0, stream>>>(ACT3, W4, ACT4, nullptr, ZP);
  // deconv: 64->32, 2x upsample via 4 parity GEMMs
  deconv_wmma<<<dim3(16, 4, ND), 256, 0, stream>>>(ACT4, WD, ACT5, ZP);
  // conv5: 32->1 (B zero-padded to 16 columns), f32 out
  conv3x3_wmma<32, 16, 1, 64, 128, false, true, 1>
      <<<dim3(64, 1, ND), 256, 0, stream>>>(ACT5, W5, nullptr, (float*)d_out, ZP);
}
